// RAGModule_18356690223140
// MI455X (gfx1250) — matrix operand order repaired and checked by reading, not compiled, not run on hardware
//
#include <hip/hip_runtime.h>
#include <stdint.h>

// ---------------------------------------------------------------------------
// Cosine-similarity brute-force kNN for MI455X (gfx1250).
//   query      : [64, 512]  fp32
//   embeddings : [500000, 512] fp32
//   k = 10
// Bandwidth-bound (1.02 GB stream, 32 FLOP/byte -> need >745 TF/s to keep up
// with 23.3 TB/s HBM) => convert to bf16 on the fly and use
// v_wmma_f32_16x16x32_bf16 with fp32 accumulation.
// ---------------------------------------------------------------------------

#define NQ      64          // queries
#define DIM     512         // embedding dim
#define ROWS    32          // embedding rows per LDS tile
#define NEMB    500000      // corpus size (divisible by ROWS)
#define NTILES  (NEMB / ROWS)   // 15625
#define NBLK    1024        // blocks for main kernel
#define KSEL    10          // top-k
#define EPSN    1e-8f

typedef __attribute__((ext_vector_type(16))) __bf16 v16bf;
typedef __attribute__((ext_vector_type(8)))  float  v8f;

union Frag {
  v16bf v;
  uint4 q[2];
};

__device__ __forceinline__ float neg_inf() { return -__builtin_inff(); }

// Sorted (descending) top-k insertion, fully unrolled -> stays in VGPRs.
__device__ __forceinline__ void topk_insert(float (&ts)[KSEL], int (&ti)[KSEL],
                                            float s, int idx) {
  if (s <= ts[KSEL - 1]) return;
  ts[KSEL - 1] = s;
  ti[KSEL - 1] = idx;
#pragma unroll
  for (int j = KSEL - 1; j > 0; --j) {
    if (ts[j] > ts[j - 1]) {
      float tf = ts[j]; ts[j] = ts[j - 1]; ts[j - 1] = tf;
      int   tn = ti[j]; ti[j] = ti[j - 1]; ti[j - 1] = tn;
    }
  }
}

// ---------------------------------------------------------------------------
// Kernel 1: normalize queries, write bf16 qn[64][512] to workspace.
// ---------------------------------------------------------------------------
__global__ __launch_bounds__(256) void qnorm_kernel(const float* __restrict__ q,
                                                    uint16_t* __restrict__ qn_raw) {
  __bf16* qn = (__bf16*)qn_raw;
  const int b   = blockIdx.x;     // query row
  const int tid = threadIdx.x;    // 256 threads, 2 elements each
  __shared__ float red[256];

  const float2 v = ((const float2*)(q + (size_t)b * DIM))[tid];
  red[tid] = v.x * v.x + v.y * v.y;
  __syncthreads();
  for (int off = 128; off > 0; off >>= 1) {
    if (tid < off) red[tid] += red[tid + off];
    __syncthreads();
  }
  const float scale = 1.0f / (sqrtf(red[0]) + EPSN);
  qn[(size_t)b * DIM + 2 * tid]     = (__bf16)(v.x * scale);
  qn[(size_t)b * DIM + 2 * tid + 1] = (__bf16)(v.y * scale);
}

// ---------------------------------------------------------------------------
// Kernel 2: streaming GEMM (bf16 WMMA) + per-block top-k candidates.
// Block = 256 threads = 8 waves. Each wave owns one 16x16 (query x row) tile:
//   wave w: qt = w>>1 (query tile 0..3), nt = w&1 (row tile 0..1).
// ---------------------------------------------------------------------------
__global__ __launch_bounds__(256) void knn_main_kernel(
    const float* __restrict__ emb, const uint16_t* __restrict__ qn_raw,
    float* __restrict__ candS, int* __restrict__ candI) {
  __shared__ __align__(16) __bf16 sE[ROWS * DIM];             // 32 KB bf16 tile
  __shared__ float sScale[ROWS];                              // 1/(||e||+eps)
  __shared__ float sRed[ROWS * 8];                            // sumsq partials
  __shared__ __align__(16) unsigned char sBuf[NQ * 4 * KSEL * 8];  // 20 KB
  float* sScores = (float*)sBuf;                              // [NQ][ROWS] (8 KB)

  const __bf16* qn = (const __bf16*)qn_raw;

  const int tid  = threadIdx.x;
  const int lane = tid & 31;
  const int wave = tid >> 5;
  const int qt   = wave >> 1;         // 0..3
  const int nt   = wave & 1;          // 0..1
  const int l15  = lane & 15;
  const int hi   = (lane < 16) ? 0 : 1;

  // top-k scan assignment: 4 threads per query, 8 rows each
  const int sq   = tid >> 2;
  const int part = tid & 3;

  // E-tile load assignment: 8 threads per row, 64 cols each
  const int erow = tid >> 3;
  const int esub = tid & 7;

  float ts[KSEL];
  int   ti[KSEL];
#pragma unroll
  for (int j = 0; j < KSEL; ++j) { ts[j] = neg_inf(); ti[j] = 0; }

  // A fragment base (ISA 16-bit A 16x32 layout): lanes 0-15 take K 0..7 &
  // 16..23, lanes 16-31 take K 8..15 & 24..31 of each 32-slice.
  const __bf16* aBase = qn + (size_t)(qt * 16 + l15) * DIM + (hi ? 8 : 0);

  for (int tile = blockIdx.x; tile < NTILES; tile += NBLK) {
    const int rowBase = tile * ROWS;

    // ---- stream 32x512 fp32 -> bf16 LDS; accumulate row sum-of-squares ----
    {
      const float4* src =
          (const float4*)(emb + (size_t)(rowBase + erow) * DIM + esub * 64);
      uint2* dst = (uint2*)(sE + erow * DIM + esub * 64);
      float ss = 0.f;
#pragma unroll
      for (int i = 0; i < 16; ++i) {
        const float4 v = src[i];
        ss += v.x * v.x + v.y * v.y + v.z * v.z + v.w * v.w;
        union { __bf16 h[4]; uint2 u; } p;
        p.h[0] = (__bf16)v.x; p.h[1] = (__bf16)v.y;
        p.h[2] = (__bf16)v.z; p.h[3] = (__bf16)v.w;
        dst[i] = p.u;                                   // ds_store_b64
      }
      sRed[erow * 8 + esub] = ss;
    }
    __syncthreads();
    if (tid < ROWS) {
      float s = 0.f;
#pragma unroll
      for (int i = 0; i < 8; ++i) s += sRed[tid * 8 + i];
      sScale[tid] = 1.0f / (sqrtf(s) + EPSN);
    }
    __syncthreads();

    // ---- 16x16x512 via 16 x v_wmma_f32_16x16x32_bf16 ----
    v8f acc = {};
    // B fragment (32x16): lane holds K for column N=l15
    // (lanes 0-15: K 0..7 & 16..23, lanes 16-31: K 8..15 & 24..31 of each 32-slice).
    const __bf16* bBase = sE + (size_t)(nt * 16 + l15) * DIM + (hi ? 8 : 0);
#pragma unroll
    for (int kk = 0; kk < DIM; kk += 32) {
      Frag a, b;
      a.q[0] = *(const uint4*)(aBase + kk);        // global (L2-hot) 16 B
      a.q[1] = *(const uint4*)(aBase + kk + 16);
      b.q[0] = *(const uint4*)(bBase + kk);        // ds_load_b128 x2
      b.q[1] = *(const uint4*)(bBase + kk + 16);
      acc = __builtin_amdgcn_wmma_f32_16x16x32_bf16(
          /*neg_a=*/false, a.v, /*neg_b=*/false, b.v,
          /*c_mod=*/(short)0, acc, /*reuse_a=*/false, /*reuse_b=*/false);
    }

    // ---- scale by 1/(||e||+eps), dump to LDS score tile ----
    {
      const int col = nt * 16 + l15;
      const float rs = sScale[col];
#pragma unroll
      for (int i = 0; i < 8; ++i) {
        const int qrow = qt * 16 + i + (hi ? 8 : 0);  // C/D layout: M = i(+8)
        sScores[qrow * ROWS + col] = acc[i] * rs;
      }
    }
    __syncthreads();

    // ---- running per-thread top-k over this tile ----
#pragma unroll
    for (int r = 0; r < 8; ++r) {
      const int row = part * 8 + r;
      topk_insert(ts, ti, sScores[sq * ROWS + row], rowBase + row);
    }
    __syncthreads();
  }

  // ---- merge the 4 sorted lists per query; emit block candidates ----
  float* mS = (float*)sBuf;                        // [NQ][4][KSEL]
  int*   mI = (int*)(sBuf + NQ * 4 * KSEL * 4);
#pragma unroll
  for (int j = 0; j < KSEL; ++j) {
    mS[(sq * 4 + part) * KSEL + j] = ts[j];
    mI[(sq * 4 + part) * KSEL + j] = ti[j];
  }
  __syncthreads();
  if (part == 0) {
    int p[4] = {0, 0, 0, 0};
    const int base = sq * 4 * KSEL;
    float* oS = candS + ((size_t)sq * NBLK + blockIdx.x) * KSEL;
    int*   oI = candI + ((size_t)sq * NBLK + blockIdx.x) * KSEL;
    for (int j = 0; j < KSEL; ++j) {
      float best = neg_inf();
      int bl = 0;
#pragma unroll
      for (int l = 0; l < 4; ++l) {
        if (p[l] < KSEL) {
          const float v = mS[base + l * KSEL + p[l]];
          if (v > best) { best = v; bl = l; }
        }
      }
      oS[j] = best;
      oI[j] = mI[base + bl * KSEL + p[bl]];
      p[bl]++;
    }
  }
}

// ---------------------------------------------------------------------------
// Kernel 3: per-query final top-10 over NBLK*10 candidates.
// ---------------------------------------------------------------------------
__global__ __launch_bounds__(256) void knn_reduce_kernel(
    const float* __restrict__ candS, const int* __restrict__ candI,
    float* __restrict__ out) {
  const int q   = blockIdx.x;
  const int tid = threadIdx.x;
  const int M   = NBLK * KSEL;                 // 10240 candidates
  __shared__ float lS[256 * KSEL];             // 10 KB
  __shared__ int   lI[256 * KSEL];             // 10 KB
  __shared__ float rS[256];
  __shared__ int   rI[256];

  float ts[KSEL];
  int   ti[KSEL];
#pragma unroll
  for (int j = 0; j < KSEL; ++j) { ts[j] = neg_inf(); ti[j] = 0; }

  for (int i = tid; i < M; i += 256)
    topk_insert(ts, ti, candS[(size_t)q * M + i], candI[(size_t)q * M + i]);

#pragma unroll
  for (int j = 0; j < KSEL; ++j) {
    lS[tid * KSEL + j] = ts[j];
    lI[tid * KSEL + j] = ti[j];
  }
  __syncthreads();

  for (int j = 0; j < KSEL; ++j) {
    float bs = neg_inf();
    int   bp = -1;
    for (int i = tid; i < 256 * KSEL; i += 256) {
      if (lS[i] > bs) { bs = lS[i]; bp = i; }
    }
    rS[tid] = bs;
    rI[tid] = bp;
    __syncthreads();
    for (int off = 128; off > 0; off >>= 1) {
      if (tid < off && rS[tid + off] > rS[tid]) {
        rS[tid] = rS[tid + off];
        rI[tid] = rI[tid + off];
      }
      __syncthreads();
    }
    if (tid == 0) {
      const int w = rI[0];
      out[q * KSEL + j] = rS[0];                                 // scores
      out[NQ * KSEL + q * KSEL + j] = (w >= 0) ? (float)lI[w] : 0.f;  // indices
      if (w >= 0) lS[w] = neg_inf();
    }
    __syncthreads();
  }
}

// ---------------------------------------------------------------------------
extern "C" void kernel_launch(void* const* d_in, const int* in_sizes, int n_in,
                              void* d_out, int out_size, void* d_ws,
                              size_t ws_size, hipStream_t stream) {
  (void)in_sizes; (void)n_in; (void)out_size; (void)ws_size;

  const float* query = (const float*)d_in[0];   // [64, 512] fp32
  const float* emb   = (const float*)d_in[1];   // [500000, 512] fp32
  // d_in[2] holds k (== 10), compiled in as KSEL.
  float* out = (float*)d_out;                   // [64*10 scores | 64*10 idx]

  // Workspace layout (needs ~5.1 MB):
  //   [0,          65536)              : qn bf16 [64][512]
  //   [65536,      65536 + 2621440)    : candS   [64][NBLK][10] f32
  //   [+2621440,   + another 2621440)  : candI   [64][NBLK][10] i32
  unsigned char* ws = (unsigned char*)d_ws;
  uint16_t* qn = (uint16_t*)ws;
  const size_t candBytes = (size_t)NQ * NBLK * KSEL * 4;
  float* candS = (float*)(ws + 65536);
  int*   candI = (int*)(ws + 65536 + candBytes);

  qnorm_kernel<<<NQ, 256, 0, stream>>>(query, qn);
  knn_main_kernel<<<NBLK, 256, 0, stream>>>(emb, qn, candS, candI);
  knn_reduce_kernel<<<NQ, 256, 0, stream>>>(candS, candI, out);
}
